// LeafBlockAttention_14731737825595
// MI455X (gfx1250) — matrix operand order repaired and checked by reading, not compiled, not run on hardware
//
#include <hip/hip_runtime.h>
#include <hip/hip_bf16.h>

#define DIM   512
#define L_    64
#define H_    8
#define HD_   64
#define KEY_  65
#define BATCH 2
#define NTOK  32768
#define NB    512            // NTOK / L
#define NBLK  (BATCH * NB)   // 1024
#define C3    (3 * DIM)      // 1536

typedef __attribute__((ext_vector_type(16))) __bf16 v16bf;
typedef __attribute__((ext_vector_type(8)))  __bf16 v8bf;
typedef __attribute__((ext_vector_type(8)))  float  v8f;
typedef int v4i __attribute__((vector_size(16)));   // matches builtin's b128 pointee

// gfx1250 async global->LDS copy path (ISA 08_async_tensor: GLOBAL_LOAD_ASYNC_TO_LDS_B128)
#if __has_builtin(__builtin_amdgcn_global_load_async_to_lds_b128)
#define USE_ASYNC_LDS 1
#else
#define USE_ASYNC_LDS 0
#endif

__device__ __forceinline__ void wait_async0() {
#if __has_builtin(__builtin_amdgcn_s_wait_asynccnt)
  __builtin_amdgcn_s_wait_asynccnt(0);
#else
  asm volatile("s_wait_asynccnt 0" ::: "memory");
#endif
}

#if USE_ASYNC_LDS
__device__ __forceinline__ void async_cp16(const __bf16* g, __bf16* l) {
  __builtin_amdgcn_global_load_async_to_lds_b128(
      (__attribute__((address_space(1))) v4i*)(g),
      (__attribute__((address_space(3))) v4i*)(l),
      0, 0);
}
#endif

// ---------------------------------------------------------------------------
// WMMA fragment loaders (wave32, v_wmma_f32_16x16x32_bf16)
// A operand: 16x32 (MxK).  lane: m = lane&15, h = lane>>4
//   a[i]   (i<8) = A[m][h*8 + i]
//   a[8+i] (i<8) = A[m][16 + h*8 + i]
// B operand: 32x16 (KxN) stored transposed in LDS as [n][k] rows.
//   lane: n = lane&15, h = lane>>4;  b[i] = Bt[n][h*8 + i], b[8+i] = Bt[n][16 + h*8 + i]
// C/D: lane: n = lane&15;  c[i] -> row m = 8*(lane>>4) + i
// ---------------------------------------------------------------------------
__device__ __forceinline__ v16bf frag_a(const __bf16* base, int stride) {
  const int lane = threadIdx.x & 31;
  const int m = lane & 15, h = lane >> 4;
  const __bf16* p = base + m * stride + h * 8;
  v16bf f;
#pragma unroll
  for (int i = 0; i < 8; ++i) f[i] = p[i];
#pragma unroll
  for (int i = 0; i < 8; ++i) f[8 + i] = p[16 + i];
  return f;
}

__device__ __forceinline__ v16bf frag_b(const __bf16* base, int stride) {
  const int lane = threadIdx.x & 31;
  const int n = lane & 15, h = lane >> 4;
  const __bf16* p = base + n * stride + h * 8;
  v16bf f;
#pragma unroll
  for (int i = 0; i < 8; ++i) f[i] = p[i];
#pragma unroll
  for (int i = 0; i < 8; ++i) f[8 + i] = p[16 + i];
  return f;
}

// ---------------------------------------------------------------------------
// Kernel 1: convert f32 weights -> bf16
// ---------------------------------------------------------------------------
__global__ void cvt_weights(const float* __restrict__ qkv_w,
                            const float* __restrict__ proj_w,
                            __bf16* __restrict__ wq, __bf16* __restrict__ wp) {
  int i = blockIdx.x * 256 + threadIdx.x;
  if (i < C3 * DIM)  wq[i] = (__bf16)qkv_w[i];
  if (i < DIM * DIM) wp[i] = (__bf16)proj_w[i];
}

// ---------------------------------------------------------------------------
// Kernel 2: per leaf block, build bf16 kv input: rows 0..63 = x, row 64 = mean
// kvx layout: (B*nb, KEY, DIM) bf16
// ---------------------------------------------------------------------------
__global__ void prep_kvx(const float* __restrict__ x, __bf16* __restrict__ kvx) {
  const int blk = blockIdx.x;                 // 0 .. NBLK-1
  const int t = threadIdx.x;                  // 0 .. 255
  const float* xb = x + (size_t)blk * L_ * DIM;
  __bf16* ob = kvx + (size_t)blk * KEY_ * DIM;
#pragma unroll
  for (int j = 0; j < 2; ++j) {
    int c = t + j * 256;
    float sum = 0.f;
    for (int r = 0; r < L_; ++r) {
      float v = xb[(size_t)r * DIM + c];
      ob[(size_t)r * DIM + c] = (__bf16)v;
      sum += v;
    }
    ob[(size_t)L_ * DIM + c] = (__bf16)(sum * (1.0f / (float)L_));
  }
}

// ---------------------------------------------------------------------------
// Kernel 3/5: tiled bf16 GEMM, out = A(M,512) @ W(N,512)^T + bias
// grid = (M/128, N/64), block = 256 threads (8 waves), wave tile 32x32.
// Async double-buffered K-loop: tile k+1 streams memory->LDS (ASYNCcnt)
// while WMMAs consume tile k; one barrier per K-step.
// ---------------------------------------------------------------------------
template <bool OUTF32>
__global__ void gemm128x64(const __bf16* __restrict__ A, const __bf16* __restrict__ W,
                           const float* __restrict__ bias, void* __restrict__ out,
                           int N) {
  __shared__ __attribute__((aligned(16))) __bf16 As[2][128 * 32];
  __shared__ __attribute__((aligned(16))) __bf16 Ws[2][64 * 32];
  const int t = threadIdx.x;
  const int mb = blockIdx.x, nb_ = blockIdx.y;
  const int wv = t >> 5, lane = t & 31;
  const int wm = wv >> 1, wn = wv & 1;
  const __bf16* Ab = A + (size_t)mb * 128 * DIM;
  const __bf16* Wb = W + (size_t)nb_ * 64 * DIM;

  // per-thread 16B chunk coordinates (A: 2 chunks, W: 1 chunk)
  const int rr = t >> 2;               // 0..63
  const int cc = (t & 3) * 8;          // 0,8,16,24

  v8f acc[2][2] = {};

#if USE_ASYNC_LDS
  // ---- async double-buffered pipeline ----
  {
    // preload tile 0
    async_cp16(&Ab[(size_t)rr * DIM + cc],        &As[0][rr * 32 + cc]);
    async_cp16(&Ab[(size_t)(64 + rr) * DIM + cc], &As[0][(64 + rr) * 32 + cc]);
    async_cp16(&Wb[(size_t)rr * DIM + cc],        &Ws[0][rr * 32 + cc]);
  }
#pragma unroll 1
  for (int it = 0; it < DIM / 32; ++it) {
    wait_async0();          // this wave's tile-it chunks have landed in LDS
    __syncthreads();        // all waves landed + all done reading buf[(it+1)&1]
    if (it + 1 < DIM / 32) {
      const int k0 = (it + 1) * 32;
      const int nb2 = (it + 1) & 1;
      async_cp16(&Ab[(size_t)rr * DIM + k0 + cc],        &As[nb2][rr * 32 + cc]);
      async_cp16(&Ab[(size_t)(64 + rr) * DIM + k0 + cc], &As[nb2][(64 + rr) * 32 + cc]);
      async_cp16(&Wb[(size_t)rr * DIM + k0 + cc],        &Ws[nb2][rr * 32 + cc]);
    }
    const __bf16* Asb = As[it & 1];
    const __bf16* Wsb = Ws[it & 1];
    v16bf a0 = frag_a(&Asb[(wm * 32 + 0) * 32], 32);
    v16bf a1 = frag_a(&Asb[(wm * 32 + 16) * 32], 32);
    v16bf b0 = frag_b(&Wsb[(wn * 32 + 0) * 32], 32);
    v16bf b1 = frag_b(&Wsb[(wn * 32 + 16) * 32], 32);
    acc[0][0] = __builtin_amdgcn_wmma_f32_16x16x32_bf16(false, a0, false, b0, (short)0, acc[0][0], false, false);
    acc[0][1] = __builtin_amdgcn_wmma_f32_16x16x32_bf16(false, a0, false, b1, (short)0, acc[0][1], false, false);
    acc[1][0] = __builtin_amdgcn_wmma_f32_16x16x32_bf16(false, a1, false, b0, (short)0, acc[1][0], false, false);
    acc[1][1] = __builtin_amdgcn_wmma_f32_16x16x32_bf16(false, a1, false, b1, (short)0, acc[1][1], false, false);
  }
#else
  // ---- fallback: synchronous staging ----
  for (int k0 = 0; k0 < DIM; k0 += 32) {
#pragma unroll
    for (int j = 0; j < 2; ++j) {
      int r = rr + j * 64;
      *(v8bf*)&As[0][r * 32 + cc] = *(const v8bf*)&Ab[(size_t)r * DIM + k0 + cc];
    }
    *(v8bf*)&Ws[0][rr * 32 + cc] = *(const v8bf*)&Wb[(size_t)rr * DIM + k0 + cc];
    if (k0 + 32 < DIM) __builtin_prefetch(&Ab[(size_t)(t >> 1) * DIM + k0 + 32], 0, 1);
    __syncthreads();
    v16bf a0 = frag_a(&As[0][(wm * 32 + 0) * 32], 32);
    v16bf a1 = frag_a(&As[0][(wm * 32 + 16) * 32], 32);
    v16bf b0 = frag_b(&Ws[0][(wn * 32 + 0) * 32], 32);
    v16bf b1 = frag_b(&Ws[0][(wn * 32 + 16) * 32], 32);
    acc[0][0] = __builtin_amdgcn_wmma_f32_16x16x32_bf16(false, a0, false, b0, (short)0, acc[0][0], false, false);
    acc[0][1] = __builtin_amdgcn_wmma_f32_16x16x32_bf16(false, a0, false, b1, (short)0, acc[0][1], false, false);
    acc[1][0] = __builtin_amdgcn_wmma_f32_16x16x32_bf16(false, a1, false, b0, (short)0, acc[1][0], false, false);
    acc[1][1] = __builtin_amdgcn_wmma_f32_16x16x32_bf16(false, a1, false, b1, (short)0, acc[1][1], false, false);
    __syncthreads();
  }
#endif

  const int n_in = lane & 15, mh = (lane >> 4) * 8;
#pragma unroll
  for (int mi = 0; mi < 2; ++mi)
#pragma unroll
    for (int ni = 0; ni < 2; ++ni) {
      int gcol = nb_ * 64 + wn * 32 + ni * 16 + n_in;
      float bv = bias[gcol];
#pragma unroll
      for (int i = 0; i < 8; ++i) {
        int grow = mb * 128 + wm * 32 + mi * 16 + mh + i;
        float v = acc[mi][ni][i] + bv;
        if (OUTF32) ((float*)out)[(size_t)grow * N + gcol] = v;
        else        ((__bf16*)out)[(size_t)grow * N + gcol] = (__bf16)v;
      }
    }
}

// ---------------------------------------------------------------------------
// Kernel 4: per (block, head) attention.
// scores = q@k^T * 1/8 (WMMA, 64x80 padded) -> bias/mask/softmax/lew (f32)
// out = combined(64x96 padded) @ v (WMMA) -> bf16 attn_out (B*nb*64, 512)
// ---------------------------------------------------------------------------
__global__ void attn_kernel(const __bf16* __restrict__ qkv,
                            const int* __restrict__ amask,
                            const float* __restrict__ ef,
                            const float* __restrict__ eg_w,
                            const float* __restrict__ eg_b,
                            __bf16* __restrict__ attn_out) {
  __shared__ __attribute__((aligned(16))) __bf16 qs[64 * 64];
  __shared__ __attribute__((aligned(16))) __bf16 ks[80 * 64];
  __shared__ __attribute__((aligned(16))) float  S[64 * 80];
  __shared__ __attribute__((aligned(16))) __bf16 cm[64 * 96];
  __shared__ __attribute__((aligned(16))) __bf16 vt[64 * 96];

  const int blk = blockIdx.x;      // 0 .. NBLK-1
  const int h = blockIdx.y;        // 0 .. 7
  const int t = threadIdx.x;       // 0 .. 127 (4 waves)
  const int wv = t >> 5, lane = t & 31;
  const __bf16* base = qkv + (size_t)blk * KEY_ * C3;

  // stage q (64x64), k^T rows (80x64, padded), v^T (64x96, padded)
  for (int e = t; e < 64 * 64; e += 128) {
    int r = e >> 6, d = e & 63;
    qs[e] = base[(size_t)r * C3 + h * 64 + d];
  }
  for (int e = t; e < 80 * 64; e += 128) {
    int r = e >> 6, d = e & 63;
    ks[e] = (r < KEY_) ? base[(size_t)r * C3 + DIM + h * 64 + d] : (__bf16)0.f;
  }
  for (int e = t; e < 64 * 96; e += 128) {
    int d = e / 96, r = e % 96;
    vt[e] = (r < KEY_) ? base[(size_t)r * C3 + 2 * DIM + h * 64 + d] : (__bf16)0.f;
  }
  __syncthreads();

  // ---- scores: each wave does a 16-row strip x 5 n-tiles, K = 64 ----
  {
    const int m0 = wv * 16;
    const int n_in = lane & 15, mh = (lane >> 4) * 8;
#pragma unroll
    for (int nt = 0; nt < 5; ++nt) {
      v8f acc = {};
#pragma unroll
      for (int kk = 0; kk < 2; ++kk) {
        v16bf a = frag_a(&qs[m0 * 64 + kk * 32], 64);
        v16bf b = frag_b(&ks[(nt * 16) * 64 + kk * 32], 64);
        acc = __builtin_amdgcn_wmma_f32_16x16x32_bf16(false, a, false, b, (short)0, acc, false, false);
      }
#pragma unroll
      for (int i = 0; i < 8; ++i)
        S[(m0 + mh + i) * 80 + nt * 16 + n_in] = acc[i] * 0.125f;  // HD^-0.5
    }
  }
  __syncthreads();

  // ---- bias + mask + softmax + lew (one thread per query row) ----
  if (t < 64) {
    const int q = t;
    const float* efr = ef + ((size_t)blk * L_ + q) * KEY_ * 4;
    const int* mr = amask + ((size_t)blk * L_ + q) * KEY_;
    const float w0 = eg_w[h * 4 + 0], w1 = eg_w[h * 4 + 1];
    const float w2 = eg_w[h * 4 + 2], w3 = eg_w[h * 4 + 3];
    const float eb = eg_b[h];

    float maxv = -3.0e38f;
    for (int k = 0; k < KEY_; ++k) {
      float s;
      if (mr[k] == 0) {
        s = -3.0e38f;
      } else {
        float b3 = (k == KEY_ - 1 || k == q) ? 1.0f : efr[k * 4 + 3];
        s = S[q * 80 + k] + b3;
      }
      S[q * 80 + k] = s;
      maxv = fmaxf(maxv, s);
    }
    float sum = 0.f;
    for (int k = 0; k < KEY_; ++k) {
      float p = __expf(S[q * 80 + k] - maxv);   // masked -> exp(-huge) = 0
      S[q * 80 + k] = p;
      sum += p;
    }
    const float inv = 1.0f / sum;
    for (int k = 0; k < KEY_; ++k) {
      float comb = 0.f;
      if (mr[k] != 0) {
        float b0, b1, b2, b3;
        if (k == KEY_ - 1 || k == q) { b0 = b1 = b2 = 0.f; b3 = 1.f; }
        else { b0 = efr[k*4]; b1 = efr[k*4+1]; b2 = efr[k*4+2]; b3 = efr[k*4+3]; }
        float lew = b0 * w0 + b1 * w1 + b2 * w2 + b3 * w3 + eb;
        comb = S[q * 80 + k] * inv + lew;
      }
      cm[q * 96 + k] = (__bf16)comb;
    }
    for (int k = KEY_; k < 96; ++k) cm[q * 96 + k] = (__bf16)0.f;
  }
  __syncthreads();

  // ---- out = combined(64x96) @ v : K padded to 96 ----
  {
    const int m0 = wv * 16;
    const int n_in = lane & 15, mh = (lane >> 4) * 8;
#pragma unroll
    for (int nt = 0; nt < 4; ++nt) {
      v8f acc = {};
#pragma unroll
      for (int kk = 0; kk < 3; ++kk) {
        v16bf a = frag_a(&cm[m0 * 96 + kk * 32], 96);
        v16bf b = frag_b(&vt[(nt * 16) * 96 + kk * 32], 96);
        acc = __builtin_amdgcn_wmma_f32_16x16x32_bf16(false, a, false, b, (short)0, acc, false, false);
      }
#pragma unroll
      for (int i = 0; i < 8; ++i) {
        int row = m0 + mh + i, col = h * 64 + nt * 16 + n_in;
        attn_out[((size_t)blk * L_ + row) * DIM + col] = (__bf16)acc[i];
      }
    }
  }
}

// ---------------------------------------------------------------------------
// Host launcher
// ---------------------------------------------------------------------------
extern "C" void kernel_launch(void* const* d_in, const int* in_sizes, int n_in,
                              void* d_out, int out_size, void* d_ws, size_t ws_size,
                              hipStream_t stream) {
  const float* x          = (const float*)d_in[0];
  const int*   attn_mask  = (const int*)d_in[1];
  const float* edge_feats = (const float*)d_in[2];
  const float* qkv_w      = (const float*)d_in[3];
  const float* qkv_b      = (const float*)d_in[4];
  const float* proj_w     = (const float*)d_in[5];
  const float* proj_b     = (const float*)d_in[6];
  const float* eg_w       = (const float*)d_in[7];
  const float* eg_b       = (const float*)d_in[8];

  char* ws = (char*)d_ws;
  // workspace layout (bytes)
  __bf16* wq     = (__bf16*)(ws + 0);           //  1,572,864  (1536x512 bf16)
  __bf16* wp     = (__bf16*)(ws + 1572864);     //    524,288  (512x512 bf16)
  __bf16* kvx    = (__bf16*)(ws + 2097152);     // 68,157,440  (1024x65x512 bf16)
  __bf16* qkv16  = (__bf16*)(ws + 70254592);    // 204,472,320 (1024x65x1536 bf16)
  __bf16* attn16 = (__bf16*)(ws + 274726912);   // 67,108,864  (65536x512 bf16)

  cvt_weights<<<dim3((C3 * DIM + 255) / 256), dim3(256), 0, stream>>>(qkv_w, proj_w, wq, wp);
  prep_kvx<<<dim3(NBLK), dim3(256), 0, stream>>>(x, kvx);
  // QKV GEMM: M = 1024*65 = 66560 = 520*128, N = 1536, K = 512
  gemm128x64<false><<<dim3(NBLK * KEY_ / 128, C3 / 64), dim3(256), 0, stream>>>(
      kvx, wq, qkv_b, (void*)qkv16, C3);
  attn_kernel<<<dim3(NBLK, H_), dim3(128), 0, stream>>>(
      qkv16, attn_mask, edge_feats, eg_w, eg_b, attn16);
  // Proj GEMM: M = 65536 = 512*128, N = 512, K = 512, f32 output
  gemm128x64<true><<<dim3(BATCH * NTOK / 128, DIM / 64), dim3(256), 0, stream>>>(
      attn16, wp, proj_b, d_out, DIM);
}
